// DecoderLSTM_54116587929803
// MI455X (gfx1250) — matrix operand order repaired and checked by reading, not compiled, not run on hardware
//
#include <hip/hip_runtime.h>
#include <stdint.h>

// CDNA5 WMMA fragment types (wave32)
typedef __attribute__((ext_vector_type(16))) __bf16 v16bf;
typedef __attribute__((ext_vector_type(8)))  float  v8f;

#define B_     256
#define T_     2048
#define VOCAB_ 7
#define EMB_   20
#define HID_   128
#define OUT_   6
#define GATES  (4 * HID_)   // 512
#define MB     16           // batch rows per workgroup

union FragU {
  v16bf v;
  uint4 q[2];
  unsigned short us[16];
};

// round-to-nearest-even f32 -> bf16 bit pattern
__device__ __forceinline__ unsigned short f2bf(float f) {
  uint32_t u = __float_as_uint(f);
  u += 0x7fffu + ((u >> 16) & 1u);
  return (unsigned short)(u >> 16);
}

__device__ __forceinline__ float fast_sigmoid(float x) {
  return __builtin_amdgcn_rcpf(1.0f + __expf(-x));
}
__device__ __forceinline__ float fast_tanh(float x) {
  return 1.0f - 2.0f * __builtin_amdgcn_rcpf(__expf(2.0f * x) + 1.0f);
}

__global__ __launch_bounds__(256, 1)
void lstm_persistent(const int*   __restrict__ inputs,  // [B,T]
                     const float* __restrict__ c0,      // [1,B,H]
                     const float* __restrict__ W_ih,    // [4H,EMB]
                     const float* __restrict__ W_hh,    // [4H,H]
                     const float* __restrict__ b_ih,    // [4H]
                     const float* __restrict__ b_hh,    // [4H]
                     const float* __restrict__ W_fc,    // [OUT,H]
                     const float* __restrict__ b_fc,    // [OUT]
                     const float* __restrict__ emb,     // [VOCAB,EMB]
                     float*       __restrict__ out)     // [B,T,OUT]
{
  __shared__ float          tbl[VOCAB_ * GATES];     // 14336 B token->gate-bias table
  __shared__ unsigned short h_lds[MB * HID_];        // 4096 B  h_{t-1} (bf16)
  __shared__ int            tokbuf[MB];
  __shared__ float          logit_lds[MB * OUT_];
  __shared__ float          bfc_lds[OUT_];

  const int tid  = threadIdx.x;
  const int lane = tid & 31;
  const int w    = tid >> 5;        // wave id 0..7
  const int half = lane >> 4;       // 0/1
  const int ln   = lane & 15;
  const int b0   = blockIdx.x * MB;
  const int jcol = (w << 4) + ln;   // hidden column this wave owns (0..127)

  // ---- preamble: x_gates table = emb @ W_ih.T + b_ih + b_hh (7x512) ----
  for (int idx = tid; idx < VOCAB_ * GATES; idx += 256) {
    int vq = idx >> 9;              // / 512
    int g  = idx & (GATES - 1);
    float a = b_ih[g] + b_hh[g];
    #pragma unroll
    for (int e = 0; e < EMB_; ++e)
      a += emb[vq * EMB_ + e] * W_ih[g * EMB_ + e];
    tbl[idx] = a;
  }
  for (int i = tid; i < MB * HID_; i += 256) h_lds[i] = 0;   // h0 = 0
  if (tid < MB)   tokbuf[tid]  = inputs[(b0 + tid) * T_];
  if (tid < OUT_) bfc_lds[tid] = b_fc[tid];

  // ---- W_hh.T as resident bf16 B-fragments: [quadrant q][ktile] ----
  // B layout (32x16 K x N, bf16): lane: n = ln, koff = half*16; VGPR v holds K=koff+2v, koff+2v+1
  FragU whh[4][4];
  #pragma unroll
  for (int qd = 0; qd < 4; ++qd) {
    int g = (w << 4) + (qd << 7) + ln;          // gate column = jcol + 128*qd
    #pragma unroll
    for (int kt = 0; kt < 4; ++kt) {
      int koff = (kt << 5) + (half << 4);
      const float* src = W_hh + g * HID_ + koff;
      #pragma unroll
      for (int i = 0; i < 16; ++i) whh[qd][kt].us[i] = f2bf(src[i]);
    }
  }
  // ---- W_fc.T B-fragments (used by wave 0), N padded 6 -> 16 with zeros ----
  FragU wfc[4];
  #pragma unroll
  for (int kt = 0; kt < 4; ++kt) {
    int koff = (kt << 5) + (half << 4);
    #pragma unroll
    for (int i = 0; i < 16; ++i)
      wfc[kt].us[i] = (ln < OUT_) ? f2bf(W_fc[ln * HID_ + koff + i]) : (unsigned short)0;
  }

  // ---- cell state resident in registers (C/D layout: m = v + half*8, n = jcol) ----
  v8f creg;
  #pragma unroll
  for (int v = 0; v < 8; ++v)
    creg[v] = c0[(size_t)(b0 + v + half * 8) * HID_ + jcol];

  __syncthreads();

  for (int t = 0; t < T_; ++t) {
    // -- phase 1: load A fragments of h_{t-1} from LDS, 16 WMMA (4 indep chains) --
    // A layout (16x32 bf16): m = ln; VGPR v: K = half*16 + v*2
    FragU af[4];
    #pragma unroll
    for (int kt = 0; kt < 4; ++kt) {
      const unsigned short* arow = h_lds + ln * HID_ + (kt << 5) + (half << 4);
      af[kt].q[0] = *(const uint4*)(arow);
      af[kt].q[1] = *(const uint4*)(arow + 8);
    }
    v8f acc[4];
    #pragma unroll
    for (int qd = 0; qd < 4; ++qd) acc[qd] = {};
    #pragma unroll
    for (int kt = 0; kt < 4; ++kt) {
      #pragma unroll
      for (int qd = 0; qd < 4; ++qd)
        acc[qd] = __builtin_amdgcn_wmma_f32_16x16x32_bf16(
            false, af[kt].v, false, whh[qd][kt].v, (short)0, acc[qd], false, false);
    }

    // -- phase 2: gate nonlinearities; acc[0..3] = i,f,g,o for (m, jcol) --
    float hnew[8];
    #pragma unroll
    for (int v = 0; v < 8; ++v) {
      int m   = v + half * 8;
      int tok = tokbuf[m];
      const float* tr = tbl + tok * GATES + jcol;
      float gi = fast_sigmoid(acc[0][v] + tr[0]);
      float gf = fast_sigmoid(acc[1][v] + tr[128]);
      float gg = fast_tanh   (acc[2][v] + tr[256]);
      float go = fast_sigmoid(acc[3][v] + tr[384]);
      float c  = gf * creg[v] + gi * gg;
      creg[v]  = c;
      hnew[v]  = go * fast_tanh(c);
    }

    __syncthreads();   // all reads of h_{t-1} / tokbuf(t) complete

    // -- phase 4: publish h_t (bf16) + prefetch tokens for t+1 --
    #pragma unroll
    for (int v = 0; v < 8; ++v)
      h_lds[(v + half * 8) * HID_ + jcol] = f2bf(hnew[v]);
    if (tid < MB) tokbuf[tid] = (t + 1 < T_) ? inputs[(b0 + tid) * T_ + t + 1] : 0;

    __syncthreads();   // h_t visible to all waves

    // -- phase 6: wave 0 fuses FC (4 WMMA) + softmax(6) + store --
    if (w == 0) {
      FragU hf[4];
      #pragma unroll
      for (int kt = 0; kt < 4; ++kt) {
        const unsigned short* arow = h_lds + ln * HID_ + (kt << 5) + (half << 4);
        hf[kt].q[0] = *(const uint4*)(arow);
        hf[kt].q[1] = *(const uint4*)(arow + 8);
      }
      v8f dlog = {};
      #pragma unroll
      for (int kt = 0; kt < 4; ++kt)
        dlog = __builtin_amdgcn_wmma_f32_16x16x32_bf16(
            false, hf[kt].v, false, wfc[kt].v, (short)0, dlog, false, false);
      if (ln < OUT_) {
        #pragma unroll
        for (int v = 0; v < 8; ++v)
          logit_lds[(v + half * 8) * OUT_ + ln] = dlog[v];
      }
      asm volatile("s_wait_dscnt 0" ::: "memory");  // in-wave LDS RAW fence
      #pragma unroll
      for (int it = 0; it < 3; ++it) {
        int idx = lane + (it << 5);         // 0..95 = 16 rows x 6 outputs
        int bm  = idx / OUT_;
        int o   = idx - bm * OUT_;
        const float* row = logit_lds + bm * OUT_;
        float v0 = row[0] + bfc_lds[0], v1 = row[1] + bfc_lds[1],
              v2 = row[2] + bfc_lds[2], v3 = row[3] + bfc_lds[3],
              v4 = row[4] + bfc_lds[4], v5 = row[5] + bfc_lds[5];
        float mx = fmaxf(fmaxf(fmaxf(v0, v1), fmaxf(v2, v3)), fmaxf(v4, v5));
        float e0 = __expf(v0 - mx), e1 = __expf(v1 - mx), e2 = __expf(v2 - mx),
              e3 = __expf(v3 - mx), e4 = __expf(v4 - mx), e5 = __expf(v5 - mx);
        float s  = e0 + e1 + e2 + e3 + e4 + e5;
        float eo = (o == 0) ? e0 : (o == 1) ? e1 : (o == 2) ? e2
                 : (o == 3) ? e3 : (o == 4) ? e4 : e5;
        out[(size_t)(b0 + bm) * (T_ * OUT_) + (size_t)t * OUT_ + o] =
            eo * __builtin_amdgcn_rcpf(s);
      }
    }
  }
}

extern "C" void kernel_launch(void* const* d_in, const int* in_sizes, int n_in,
                              void* d_out, int out_size, void* d_ws, size_t ws_size,
                              hipStream_t stream) {
  (void)in_sizes; (void)n_in; (void)out_size; (void)d_ws; (void)ws_size;
  const int*   inputs = (const int*)  d_in[0];
  const float* c0     = (const float*)d_in[1];
  const float* W_ih   = (const float*)d_in[2];
  const float* W_hh   = (const float*)d_in[3];
  const float* b_ih   = (const float*)d_in[4];
  const float* b_hh   = (const float*)d_in[5];
  const float* W_fc   = (const float*)d_in[6];
  const float* b_fc   = (const float*)d_in[7];
  const float* emb    = (const float*)d_in[8];
  float* out = (float*)d_out;

  hipLaunchKernelGGL(lstm_persistent, dim3(B_ / MB), dim3(256), 0, stream,
                     inputs, c0, W_ih, W_hh, b_ih, b_hh, W_fc, b_fc, emb, out);
}
